// GAT_38989713113447
// MI455X (gfx1250) — matrix operand order repaired and checked by reading, not compiled, not run on hardware
//
#include <hip/hip_runtime.h>
#include <hip/hip_bf16.h>
#include <cmath>
#include <cstdint>

// ---------------------------------------------------------------------------
// GAT forward for MI455X (gfx1250, wave32, WMMA).
// Heavy matmuls (H@W and att@Wh) run on v_wmma_f32_16x16x32_f16 (f16 in, f32
// acc). Softmax / logits stay f32 VALU. Attention is rank-1 (f1_i + f2_j), so
// there is no QK^T matmul; P tiles are built in registers and fed to WMMA.
// ---------------------------------------------------------------------------

typedef _Float16 f16;
typedef __attribute__((ext_vector_type(16))) _Float16 v16h;
typedef __attribute__((ext_vector_type(8)))  _Float16 v8h;
typedef __attribute__((ext_vector_type(8)))  float    v8f;

#define NNODES 3072
#define NFEAT  256
#define NHEADS 8
#define HID    128
#define OUTD   64
#define ALPHA  0.2f
#define NEGFILL (-9e15f)

__device__ __forceinline__ v16h cat16(v8h lo, v8h hi) {
  return __builtin_shufflevector(lo, hi, 0,1,2,3,4,5,6,7,8,9,10,11,12,13,14,15);
}

__device__ __forceinline__ float eluf(float v) {
  return v > 0.f ? v : (__expf(v) - 1.f);
}

// --------------------------- h = x + seed*theta, f32 -> f16 -----------------
__global__ void k_prep_h(const float* __restrict__ x, const int* __restrict__ obs,
                         const float* __restrict__ theta, f16* __restrict__ hA) {
  int row = blockIdx.x, c = threadIdx.x;           // 3072 blocks x 256 threads
  float v = x[(size_t)row * NFEAT + c];
  if (obs[row] == 1) v += theta[c];
  hA[(size_t)row * NFEAT + c] = (f16)v;
}

// --------------------------- W [K,d] f32  ->  W^T [d,K] f16 -----------------
__global__ void k_w_to_f16T(const float* __restrict__ W, f16* __restrict__ WT,
                            int K, int d) {
  size_t per = (size_t)K * d;
  const float* Ws = W + blockIdx.y * per;
  f16* WTs = WT + blockIdx.y * per;
  int idx = blockIdx.x * 256 + threadIdx.x;
  if (idx < K * d) {
    int k = idx / d, c = idx - k * d;
    WTs[(size_t)c * K + k] = (f16)Ws[idx];
  }
}

// --------------------------- WMMA GEMM: C[M,DC] = A[M,K] @ B[K,DC] ----------
// A: f16 row-major (per-head stride aStride, leading dim lda)
// BT: f16 [DC,K] row-major (pre-transposed weights, per-head stride bStride)
// Outputs: C32 f32 [M,DC] row-major, CT16 f16 [DC,M] (B operand for att@Wh).
template<int DC>
__global__ void k_wmma_gemm(const f16* __restrict__ A, size_t aStride, int lda,
                            const f16* __restrict__ BT, size_t bStride,
                            float* __restrict__ C32, size_t cStride,
                            f16* __restrict__ CT16, size_t ctStride,
                            int M, int K) {
  constexpr int CT  = DC / 16;           // column tiles per block row
  constexpr int RPB = (8 / CT) * 16;     // rows per block (8 waves / block)
  int wave = threadIdx.x >> 5, lane = threadIdx.x & 31;
  int half = lane >> 4, r = lane & 15;
  int mBase = blockIdx.x * RPB + (wave / CT) * 16;
  int nBase = (wave % CT) * 16;
  int head = blockIdx.y;

  const f16* arow = A + (size_t)head * aStride + (size_t)(mBase + r) * lda;
  const f16* brow = BT + (size_t)head * bStride + (size_t)(nBase + r) * K;

  v8f acc = {};
  for (int k = 0; k < K; k += 32) {
    // A 16x32 f16 fragment: lane row = mBase+r; K = k + 8*half + {0..7, 16..23}
    v8h alo = *(const v8h*)(arow + k + 8 * half);
    v8h ahi = *(const v8h*)(arow + k + 8 * half + 16);
    // B 32x16 f16 fragment: lane col = nBase+r; K = k + 8*half + {0..7, 16..23}
    v8h blo = *(const v8h*)(brow + k + 8 * half);
    v8h bhi = *(const v8h*)(brow + k + 8 * half + 16);
    acc = __builtin_amdgcn_wmma_f32_16x16x32_f16(
        false, cat16(alo, ahi), false, cat16(blo, bhi), (short)0, acc, false, false);
  }

  float* Cs = C32 + (size_t)head * cStride;
  f16* CTs = CT16 + (size_t)head * ctStride;
#pragma unroll
  for (int j = 0; j < 8; ++j) {          // C/D: VGPR j -> row j+8*half, col r
    int row = mBase + j + 8 * half;
    float v = acc[j];
    Cs[(size_t)row * DC + nBase + r] = v;
    CTs[(size_t)(nBase + r) * M + row] = (f16)v;
  }
}

// --------------------------- f1/f2: per-row dots with 'a' -------------------
__global__ void k_scores(const float* __restrict__ Wh, size_t whStride,
                         const float* __restrict__ a, int aStride2d,
                         float* __restrict__ f1, float* __restrict__ f2, int d) {
  int wave = threadIdx.x >> 5, lane = threadIdx.x & 31;
  int row = blockIdx.x * 4 + wave;       // 4 waves/block, one row per wave
  int head = blockIdx.y;
  const float* wr = Wh + (size_t)head * whStride + (size_t)row * d;
  const float* ah = a + (size_t)head * aStride2d;
  float s1 = 0.f, s2 = 0.f;
  for (int c = lane; c < d; c += 32) {
    float v = wr[c];
    s1 += v * ah[c];
    s2 += v * ah[d + c];
  }
  for (int off = 16; off; off >>= 1) {
    s1 += __shfl_xor(s1, off, 32);
    s2 += __shfl_xor(s2, off, 32);
  }
  if (lane == 0) {
    f1[(size_t)head * NNODES + row] = s1;
    f2[(size_t)head * NNODES + row] = s2;
  }
}

// --------------------------- row softmax stats (max, 1/sum) -----------------
__global__ void k_rowstats(const float* __restrict__ adj,
                           const float* __restrict__ f1, const float* __restrict__ f2,
                           float* __restrict__ mOut, float* __restrict__ rlOut) {
  int wave = threadIdx.x >> 5, lane = threadIdx.x & 31;
  int row = blockIdx.x * 4 + wave;
  size_t ho = (size_t)blockIdx.y * NNODES;
  float f1i = f1[ho + row];
  const float* arow = adj + (size_t)row * NNODES;
  float m = -INFINITY, l = 0.f;
  for (int j = lane; j < NNODES; j += 32) {
    __builtin_prefetch(arow + j + 512, 0, 1);
    float s = f1i + f2[ho + j];
    s = s > 0.f ? s : ALPHA * s;
    s = arow[j] > 0.f ? s : NEGFILL;     // masked fill, like the reference
    if (s > m) { l = l * __expf(m - s) + 1.f; m = s; }
    else       { l += __expf(s - m); }
  }
  for (int off = 16; off; off >>= 1) {   // merge (m,l) across lanes
    float mo = __shfl_xor(m, off, 32), lo = __shfl_xor(l, off, 32);
    float mn = fmaxf(m, mo);
    l = l * __expf(m - mn) + lo * __expf(mo - mn);
    m = mn;
  }
  if (lane == 0) { mOut[ho + row] = m; rlOut[ho + row] = 1.f / l; }
}

// --------------------------- out = elu(softmax(P) @ Wh) ---------------------
// P tiles built in registers (1/l folded into P: P rows are lane-indexed).
// WhT: f16 [DC, NNODES]. out16 (f16) or out32 (f32) destinations.
template<int DC>
__global__ void k_aggregate(const float* __restrict__ adj,
                            const float* __restrict__ f1, const float* __restrict__ f2,
                            const float* __restrict__ mIn, const float* __restrict__ rlIn,
                            const f16* __restrict__ WhT, size_t whtStride,
                            f16* __restrict__ out16, size_t o16HeadStride,
                            int ld16, int headColMul,
                            float* __restrict__ out32) {
  int wave = threadIdx.x >> 5, lane = threadIdx.x & 31;   // 32*(DC/16) threads
  int half = lane >> 4, r = lane & 15;
  int head = blockIdx.y;
  int i0 = blockIdx.x * 16;
  int nBase = wave * 16;
  size_t ho = (size_t)head * NNODES;

  int irow = i0 + r;                      // P row for this lane (both halves)
  float f1i = f1[ho + irow];
  float mi  = mIn[ho + irow];
  float rli = rlIn[ho + irow];
  const float* arow = adj + (size_t)irow * NNODES;
  const float* frow = f2 + ho;
  const f16* brow = WhT + (size_t)head * whtStride + (size_t)(nBase + r) * NNODES;

  v8f acc = {};
  for (int j0 = 0; j0 < NNODES; j0 += 32) {
    __builtin_prefetch(arow + j0 + 256, 0, 1);
    union { v16h v; f16 h[16]; } pa;
    const float* ap = arow + j0 + 8 * half;
    const float* fp = frow + j0 + 8 * half;
#pragma unroll
    for (int e = 0; e < 16; ++e) {        // A-layout K: {0..7} then {16..23}
      int off = (e < 8) ? e : (e + 8);
      float s = f1i + fp[off];
      s = s > 0.f ? s : ALPHA * s;
      s = ap[off] > 0.f ? s : NEGFILL;
      pa.h[e] = (f16)(__expf(s - mi) * rli);   // in [0,1]: f16-safe
    }
    v8h blo = *(const v8h*)(brow + j0 + 8 * half);
    v8h bhi = *(const v8h*)(brow + j0 + 8 * half + 16);
    acc = __builtin_amdgcn_wmma_f32_16x16x32_f16(
        false, pa.v, false, cat16(blo, bhi), (short)0, acc, false, false);
  }

#pragma unroll
  for (int j = 0; j < 8; ++j) {
    int row = i0 + j + 8 * half;
    int col = nBase + r;
    float v = eluf(acc[j]);
    if (out32) out32[(size_t)row * DC + col] = v;
    else out16[(size_t)head * o16HeadStride + (size_t)row * ld16 + head * headColMul + col] = (f16)v;
  }
}

// ---------------------------------------------------------------------------
extern "C" void kernel_launch(void* const* d_in, const int* in_sizes, int n_in,
                              void* d_out, int out_size, void* d_ws, size_t ws_size,
                              hipStream_t stream) {
  (void)in_sizes; (void)n_in; (void)out_size; (void)ws_size;
  const float* x     = (const float*)d_in[0];
  const float* adj   = (const float*)d_in[1];
  const int*   obs   = (const int*)  d_in[2];
  /* d_in[3] s_mat unused (method='base') */
  const float* theta = (const float*)d_in[4];
  const float* W0    = (const float*)d_in[5];
  const float* a0    = (const float*)d_in[6];
  const float* W1    = (const float*)d_in[7];
  const float* a1    = (const float*)d_in[8];
  const float* Wo    = (const float*)d_in[9];
  const float* ao    = (const float*)d_in[10];
  float* out = (float*)d_out;

  char* ws = (char*)d_ws;
  size_t off = 0;
  auto alloc = [&](size_t bytes) -> char* {
    char* p = ws + off;
    off = (off + bytes + 255) & ~(size_t)255;
    return p;
  };
  f16*   hA    = (f16*)  alloc((size_t)NNODES * NFEAT * 2);            // 1.5 MB
  f16*   W0T   = (f16*)  alloc((size_t)NHEADS * NFEAT * HID * 2);      // 0.5 MB
  f16*   W1T   = (f16*)  alloc((size_t)NHEADS * HID * HID * 2);        // 0.25 MB
  f16*   WoT   = (f16*)  alloc((size_t)NHEADS * HID * OUTD * 2);       // 0.125 MB
  float* Wh32  = (float*)alloc((size_t)NHEADS * NNODES * HID * 4);     // 12 MB (reused/stage)
  f16*   WhT16 = (f16*)  alloc((size_t)NHEADS * NNODES * HID * 2);     // 6 MB  (reused/stage)
  float* f1    = (float*)alloc((size_t)NHEADS * NNODES * 4);
  float* f2    = (float*)alloc((size_t)NHEADS * NNODES * 4);
  float* mR    = (float*)alloc((size_t)NHEADS * NNODES * 4);
  float* rlR   = (float*)alloc((size_t)NHEADS * NNODES * 4);
  f16*   h0    = (f16*)  alloc((size_t)NHEADS * NNODES * HID * 2);     // 6 MB
  f16*   hc    = (f16*)  alloc((size_t)NNODES * NHEADS * HID * 2);     // 6 MB

  const size_t sNH = (size_t)NNODES * HID;

  // ---- prep: merged input + f16 weight transposes
  k_prep_h<<<NNODES, NFEAT, 0, stream>>>(x, obs, theta, hA);
  k_w_to_f16T<<<dim3((NFEAT * HID + 255) / 256, NHEADS), 256, 0, stream>>>(W0, W0T, NFEAT, HID);
  k_w_to_f16T<<<dim3((HID * HID + 255) / 256, NHEADS), 256, 0, stream>>>(W1, W1T, HID, HID);
  k_w_to_f16T<<<dim3((NHEADS * HID * OUTD + 255) / 256, 1), 256, 0, stream>>>(Wo, WoT, NHEADS * HID, OUTD);

  // ---- layer 0 (8 heads, ELU, concat into per-head h0)
  k_wmma_gemm<HID><<<dim3(NNODES / 16, NHEADS), 256, 0, stream>>>(
      hA, 0, NFEAT, W0T, (size_t)NFEAT * HID, Wh32, sNH, WhT16, sNH, NNODES, NFEAT);
  k_scores<<<dim3(NNODES / 4, NHEADS), 128, 0, stream>>>(Wh32, sNH, a0, 2 * HID, f1, f2, HID);
  k_rowstats<<<dim3(NNODES / 4, NHEADS), 128, 0, stream>>>(adj, f1, f2, mR, rlR);
  k_aggregate<HID><<<dim3(NNODES / 16, NHEADS), 256, 0, stream>>>(
      adj, f1, f2, mR, rlR, WhT16, sNH, h0, sNH, HID, 0, nullptr);

  // ---- layer 1 (8 heads, ELU, writes directly into hc[3072,1024] at head*128)
  k_wmma_gemm<HID><<<dim3(NNODES / 16, NHEADS), 256, 0, stream>>>(
      h0, sNH, HID, W1T, (size_t)HID * HID, Wh32, sNH, WhT16, sNH, NNODES, HID);
  k_scores<<<dim3(NNODES / 4, NHEADS), 128, 0, stream>>>(Wh32, sNH, a1, 2 * HID, f1, f2, HID);
  k_rowstats<<<dim3(NNODES / 4, NHEADS), 128, 0, stream>>>(adj, f1, f2, mR, rlR);
  k_aggregate<HID><<<dim3(NNODES / 16, NHEADS), 256, 0, stream>>>(
      adj, f1, f2, mR, rlR, WhT16, sNH, hc, 0, NHEADS * HID, HID, nullptr);

  // ---- output attention layer (single head, d=64), final elu -> d_out (f32)
  k_wmma_gemm<OUTD><<<dim3(NNODES / 32, 1), 256, 0, stream>>>(
      hc, 0, NHEADS * HID, WoT, 0, Wh32, 0, WhT16, 0, NNODES, NHEADS * HID);
  k_scores<<<dim3(NNODES / 4, 1), 128, 0, stream>>>(Wh32, 0, ao, 2 * OUTD, f1, f2, OUTD);
  k_rowstats<<<dim3(NNODES / 4, 1), 128, 0, stream>>>(adj, f1, f2, mR, rlR);
  k_aggregate<OUTD><<<dim3(NNODES / 16, 1), 128, 0, stream>>>(
      adj, f1, f2, mR, rlR, WhT16, 0, nullptr, 0, 0, 0, out);
}
